// GCNConv_26663156974292
// MI455X (gfx1250) — matrix operand order repaired and checked by reading, not compiled, not run on hardware
//
#include <hip/hip_runtime.h>

typedef __attribute__((ext_vector_type(16))) __bf16 v16bf;
typedef __attribute__((ext_vector_type(8)))  __bf16 v8bf;
typedef __attribute__((ext_vector_type(8)))  float  v8f;

#define N_NODES 16384
#define D_IN    512
#define D_OUT   256
#define NT      (N_NODES / 32)   // 512 K-steps in stage 2

// ---- CDNA5 async-to-LDS path (ASYNCcnt-tracked DMA), guarded for portability
#if __has_builtin(__builtin_amdgcn_global_load_async_to_lds_b128)
#define USE_ASYNC_LDS 1
typedef int v4i __attribute__((vector_size(4 * sizeof(int))));
typedef __attribute__((address_space(1))) v4i GV4i;   // global src
typedef __attribute__((address_space(3))) v4i LV4i;   // LDS dst
#else
#define USE_ASYNC_LDS 0
#endif

#if __has_builtin(__builtin_amdgcn_s_wait_asynccnt)
#define WAIT_ASYNCCNT0() __builtin_amdgcn_s_wait_asynccnt(0)
#else
#define WAIT_ASYNCCNT0() asm volatile("s_wait_asynccnt 0x0" ::: "memory")
#endif

// =====================================================================
// Stage 1:  Ht[c][n] = (X @ W + b)^T in bf16  (4.3 GFLOP, negligible)
// =====================================================================
__global__ __launch_bounds__(256) void gcn_proj_kernel(
    const float* __restrict__ X,
    const float* __restrict__ W,
    const float* __restrict__ Bv,
    __bf16* __restrict__ Ht)
{
    __shared__ __bf16 Xs[16][40];

    const int tid    = threadIdx.x;
    const int wave   = tid >> 5;
    const int lane   = tid & 31;
    const int lane16 = lane & 15;
    const int laneHi = lane >> 4;
    const int nbase  = blockIdx.x * 16;
    const int c0     = wave * 32;

    v8f acc[2];
    acc[0] = (v8f){}; acc[1] = (v8f){};

    const int xr = tid >> 4;
    const int xk = (tid & 15) * 2;

    for (int kt = 0; kt < D_IN; kt += 32) {
        __syncthreads();
        const float* xp = X + (size_t)(nbase + xr) * D_IN + kt + xk;
        Xs[xr][xk]     = (__bf16)xp[0];
        Xs[xr][xk + 1] = (__bf16)xp[1];
        __syncthreads();

        v8bf alo = *(const v8bf*)&Xs[lane16][laneHi * 16];
        v8bf ahi = *(const v8bf*)&Xs[lane16][laneHi * 16 + 8];
        v16bf af = __builtin_shufflevector(alo, ahi,
                    0,1,2,3,4,5,6,7,8,9,10,11,12,13,14,15);

        #pragma unroll
        for (int t = 0; t < 2; ++t) {
            const int c  = c0 + t * 16 + lane16;
            const int kb = kt + laneHi * 16;
            v16bf bfr;
            #pragma unroll
            for (int j = 0; j < 16; ++j)
                bfr[j] = (__bf16)W[(size_t)(kb + j) * D_OUT + c];
            acc[t] = __builtin_amdgcn_wmma_f32_16x16x32_bf16(
                         false, af, false, bfr, (short)0, acc[t], false, false);
        }
    }

    #pragma unroll
    for (int t = 0; t < 2; ++t) {
        const int c = c0 + t * 16 + lane16;
        const float bias = Bv[c];
        #pragma unroll
        for (int r = 0; r < 8; ++r) {
            const int n = nbase + laneHi * 8 + r;
            Ht[(size_t)c * N_NODES + n] = (__bf16)(acc[t][r] + bias);
        }
    }
}

// =====================================================================
// Stage 2:  Out = A_hat @ H  (137 GFLOP; HBM-bound: 1 GiB A_hat @ 23.3 TB/s)
//   64x256 output block / workgroup; single-barrier ping-pong pipeline:
//   A_hat tile prefetched to VGPRs (cvt fp32->bf16), Ht tile via
//   async-to-LDS DMA (ASYNCcnt), 8 WMMA per wave per K-step of 32.
// =====================================================================
__global__ __launch_bounds__(256) void gcn_agg_kernel(
    const float*  __restrict__ A,    // [N, N] f32
    const __bf16* __restrict__ Ht,   // [256, N] bf16 (H transposed)
    float* __restrict__ Out)         // [N, 256] f32
{
    __shared__ __bf16 As[2][64][40];    // double-buffered 64x32 A-tile
    __shared__ __bf16 Bs[2][256][40];   // double-buffered 256x32 B-tile

    const int tid    = threadIdx.x;
    const int wave   = tid >> 5;
    const int lane   = tid & 31;
    const int lane16 = lane & 15;
    const int laneHi = lane >> 4;

    const int row0   = blockIdx.x * 64;
    const int m_tile = wave >> 1;
    const int c_base = (wave & 1) * 128;

    v8f acc[8];
    #pragma unroll
    for (int t = 0; t < 8; ++t) acc[t] = (v8f){};

    const int ar = tid >> 2;          // 0..63
    const int ak = (tid & 3) * 8;     // 0,8,16,24
    const float*  aptr = A  + (size_t)(row0 + ar) * N_NODES + ak;
    const __bf16* hptr = Ht + (size_t)tid * N_NODES;

    float4 ra0, ra1;                  // in-flight A_hat tile (next K-step)

    auto loadA = [&](int kt) {
        const float* p = aptr + kt;
        ra0 = *(const float4*)p;
        ra1 = *(const float4*)(p + 4);
    };
    auto storeA = [&](int buf) {
        v8bf pk;
        pk[0] = (__bf16)ra0.x; pk[1] = (__bf16)ra0.y;
        pk[2] = (__bf16)ra0.z; pk[3] = (__bf16)ra0.w;
        pk[4] = (__bf16)ra1.x; pk[5] = (__bf16)ra1.y;
        pk[6] = (__bf16)ra1.z; pk[7] = (__bf16)ra1.w;
        *(v8bf*)&As[buf][ar][ak] = pk;
    };
    auto issueB = [&](int kt, int buf) {
#if USE_ASYNC_LDS
        const __bf16* gp = hptr + kt;
        __bf16* lp = &Bs[buf][tid][0];
        #pragma unroll
        for (int q = 0; q < 4; ++q)
            __builtin_amdgcn_global_load_async_to_lds_b128(
                (GV4i*)(uintptr_t)(gp + 8 * q),
                (LV4i*)(unsigned)(uintptr_t)(lp + 8 * q), 0, 0);
#else
        const uint4* hp  = (const uint4*)(hptr + kt);
        uint4*       bsp = (uint4*)&Bs[buf][tid][0];
        bsp[0] = hp[0]; bsp[1] = hp[1]; bsp[2] = hp[2]; bsp[3] = hp[3];
#endif
    };

    // ---- prologue: stage tile 0
    loadA(0);
    issueB(0, 0);
    storeA(0);
#if USE_ASYNC_LDS
    WAIT_ASYNCCNT0();
#endif

    for (int i = 0; i < NT; ++i) {
        const int p = i & 1;
        __syncthreads();   // fence drains DScnt: buffer 1-p is quiescent now

        const bool more = (i + 1) < NT;
        if (more) {
            issueB((i + 1) * 32, 1 - p);   // async DMA into other buffer
            loadA((i + 1) * 32);           // A_hat prefetch into VGPRs
        }

        // ---- compute on buffer p
        const int arow = m_tile * 16 + lane16;
        v8bf alo = *(const v8bf*)&As[p][arow][laneHi * 16];
        v8bf ahi = *(const v8bf*)&As[p][arow][laneHi * 16 + 8];
        v16bf af = __builtin_shufflevector(alo, ahi,
                    0,1,2,3,4,5,6,7,8,9,10,11,12,13,14,15);

        #pragma unroll
        for (int t = 0; t < 8; ++t) {
            const int c = c_base + t * 16 + lane16;
            const v8bf* bp = (const v8bf*)&Bs[p][c][laneHi * 16];
            v16bf bfr = __builtin_shufflevector(bp[0], bp[1],
                         0,1,2,3,4,5,6,7,8,9,10,11,12,13,14,15);
            acc[t] = __builtin_amdgcn_wmma_f32_16x16x32_bf16(
                         false, af, false, bfr, (short)0, acc[t], false, false);
        }

        if (more) {
            storeA(1 - p);      // cvt + ds_store of prefetched A tile
#if USE_ASYNC_LDS
            WAIT_ASYNCCNT0();   // Bs[1-p] DMA complete before next barrier
#endif
        }
    }

    // ---- write 64x256 fp32 block
    #pragma unroll
    for (int t = 0; t < 8; ++t) {
        const int c = c_base + t * 16 + lane16;
        #pragma unroll
        for (int r = 0; r < 8; ++r) {
            const int m = row0 + m_tile * 16 + laneHi * 8 + r;
            Out[(size_t)m * D_OUT + c] = acc[t][r];
        }
    }
}

extern "C" void kernel_launch(void* const* d_in, const int* in_sizes, int n_in,
                              void* d_out, int out_size, void* d_ws, size_t ws_size,
                              hipStream_t stream) {
    const float* X    = (const float*)d_in[0];   // [16384, 512]
    const float* Ahat = (const float*)d_in[1];   // [16384, 16384]
    const float* W    = (const float*)d_in[2];   // [512, 256]
    const float* Bv   = (const float*)d_in[3];   // [256]
    float*  Out = (float*)d_out;                 // [16384, 256]
    __bf16* Ht  = (__bf16*)d_ws;                 // [256, 16384] bf16, 8 MB scratch

    gcn_proj_kernel<<<N_NODES / 16, 256, 0, stream>>>(X, W, Bv, Ht);
    gcn_agg_kernel <<<N_NODES / 64, 256, 0, stream>>>(Ahat, Ht, Out);
}
